// SimpleSNN_60473139528346
// MI455X (gfx1250) — matrix operand order repaired and checked by reading, not compiled, not run on hardware
//
#include <hip/hip_runtime.h>

typedef __attribute__((ext_vector_type(16))) __bf16 v16bf;
typedef __attribute__((ext_vector_type(8)))  __bf16 v8bf;
typedef __attribute__((ext_vector_type(8)))  float  v8f;
typedef __attribute__((ext_vector_type(4)))  unsigned int u32x4;
typedef __attribute__((ext_vector_type(8)))  int i32x8;
typedef __attribute__((ext_vector_type(4)))  int i32x4;

#define SNN_BETA   0.95f
#define SNN_THRESH 1.0f

#if defined(__AMDGCN__) && __has_builtin(__builtin_amdgcn_tensor_load_to_lds) && \
    __has_builtin(__builtin_amdgcn_s_wait_tensorcnt)
#define SNN_TDM 1
#else
#define SNN_TDM 0
#endif

__device__ __forceinline__ v16bf frag16(const __bf16* p0, const __bf16* p1) {
    v8bf lo = *(const v8bf*)p0;
    v8bf hi = *(const v8bf*)p1;
    return __builtin_shufflevector(lo, hi, 0,1,2,3,4,5,6,7,8,9,10,11,12,13,14,15);
}

#if SNN_TDM
// One 2-D TDM tile DMA: tile_h rows x tile_w bf16 elements, global row stride
// row_stride_elems, written compactly to LDS at lds_dst. Tensor dims == tile
// dims (no OOB path). D# layout per CDNA5 ISA 8.3/8.4.
__device__ __forceinline__ void tdm_load_tile_2d(
    const __bf16* gsrc, __bf16* lds_dst,
    unsigned tile_w, unsigned tile_h, unsigned row_stride_elems)
{
    const unsigned long long ga = (unsigned long long)(uintptr_t)gsrc;
    const unsigned lds_off = (unsigned)(uintptr_t)lds_dst; // addr[31:0] = LDS byte offset
    u32x4 g0 = {
        1u,                                                  // count=1 (user D#)
        lds_off,                                             // lds_addr
        (unsigned)ga,                                        // global_addr[31:0]
        (unsigned)((ga >> 32) & 0x1FFFFFFull) | (2u << 30)   // global_addr[56:32] | type=2
    };
    i32x8 g1 = {
        (int)(1u << 16),                                     // wg_mask=0, data_size=1 (2B)
        (int)((tile_w & 0xFFFFu) << 16),                     // tensor_dim0[15:0]
        (int)(((tile_w >> 16) & 0xFFFFu) | ((tile_h & 0xFFFFu) << 16)), // dim0_hi | dim1_lo
        (int)(((tile_h >> 16) & 0xFFFFu) | ((tile_w & 0xFFFFu) << 16)), // dim1_hi | tile_dim0
        (int)(tile_h & 0xFFFFu),                             // tile_dim1 | tile_dim2=0
        (int)row_stride_elems,                               // tensor_dim0_stride[31:0]
        0,                                                   // stride_hi=0 | dim1_stride_lo=0
        0
    };
    i32x4 gz = {0, 0, 0, 0};
#if defined(__clang_major__) && (__clang_major__ >= 23)
    i32x8 gz8 = {0, 0, 0, 0, 0, 0, 0, 0};
    __builtin_amdgcn_tensor_load_to_lds(g0, g1, gz, gz, gz8, 0);
#else
    __builtin_amdgcn_tensor_load_to_lds(g0, g1, gz, gz, 0);
#endif
}
#endif

// C[M,N] = A[M,K] * W[N,K]^T, fused LIF:
//   mem = beta*mem + (C + bias) - spk_prev*thresh ; spk = (mem - thresh) > 0
// 256 threads (8 wave32), 128x128 C tile, K staged 64-wide through LDS.
// TDM path: 3-buffer rotation, wave 0 keeps 2 stages of tensor_load_to_lds in
// flight (TENSORcnt in-order => s_wait_tensorcnt 2 completes only the oldest).
// Fallback: 2-buffer register-prefetch global->LDS.
template<bool ACCUM_OUT>
__global__ __launch_bounds__(256) __attribute__((amdgpu_waves_per_eu(1)))
void gemm_lif_kernel(
    const __bf16* __restrict__ A, int lda,
    const __bf16* __restrict__ W,     // [N][K] row-major
    const float*  __restrict__ bias,  // [N]
    float*        __restrict__ memv,  // [M][N] membrane state
    __bf16*       __restrict__ spk,   // [M][N] in: prev spikes, out: new spikes
    float*        __restrict__ out,   // [M][N] accumulated output (ACCUM_OUT)
    int K, int N)
{
    constexpr int KSTEP = 64;
    constexpr int NBUF  = SNN_TDM ? 3 : 2;
    constexpr int SLAB  = 2 * 128 * KSTEP;            // A tile + W tile, elems
    __shared__ __align__(16) __bf16 smem[NBUF * SLAB]; // 96 KB (TDM) / 64 KB

    const int tid  = threadIdx.x;
    const int lane = tid & 31;
    const int wave = tid >> 5;
    const int wm   = wave >> 2;      // 0..1 (M)
    const int wn   = wave & 3;       // 0..3 (N)
    const int hl   = lane >> 4;
    const int l16  = lane & 15;
    const int bm0  = blockIdx.y * 128;
    const int bn0  = blockIdx.x * 128;

    const int nstages = K / KSTEP;

#if SNN_TDM
    const bool loader = (wave == 0);
    auto stage_load = [&](int st, int p) {
        const int k0 = st * KSTEP;
        __bf16* As = smem + p * SLAB;
        __bf16* Ws = As + 128 * KSTEP;
        tdm_load_tile_2d(A + (size_t)bm0 * (size_t)lda + k0, As, KSTEP, 128, (unsigned)lda);
        tdm_load_tile_2d(W + (size_t)bn0 * (size_t)K   + k0, Ws, KSTEP, 128, (unsigned)K);
    };
    if (loader) {
        stage_load(0, 0);
        if (nstages > 1) stage_load(1, 1);
    }
    // Non-loader waves have TENSORcnt==0; wait is a no-op there (no exec branch).
    if (nstages > 1) __builtin_amdgcn_s_wait_tensorcnt(2);
    else             __builtin_amdgcn_s_wait_tensorcnt(0);
    __syncthreads();
#else
    const int crow = tid >> 3;
    const int ckc  = (tid & 7) * 8;
    const __bf16* Ag = A + (size_t)(bm0 + crow) * (size_t)lda + ckc;
    const __bf16* Wg = W + (size_t)(bn0 + crow) * (size_t)K   + ckc;
    uint4 ar[4], wr[4];
    auto ldg = [&](int k0) {
#pragma unroll
        for (int i = 0; i < 4; ++i) {
            ar[i] = *(const uint4*)(Ag + (size_t)(32 * i) * (size_t)lda + k0);
            wr[i] = *(const uint4*)(Wg + (size_t)(32 * i) * (size_t)K   + k0);
        }
    };
    auto sts = [&](int p) {
        __bf16* As = smem + p * SLAB;
        __bf16* Ws = As + 128 * KSTEP;
#pragma unroll
        for (int i = 0; i < 4; ++i) {
            *(uint4*)(As + (crow + 32 * i) * KSTEP + ckc) = ar[i];
            *(uint4*)(Ws + (crow + 32 * i) * KSTEP + ckc) = wr[i];
        }
    };
    ldg(0);
    sts(0);
    __syncthreads();
#endif

    v8f acc[4][2];
#pragma unroll
    for (int i = 0; i < 4; ++i)
#pragma unroll
        for (int j = 0; j < 2; ++j)
#pragma unroll
            for (int r = 0; r < 8; ++r)
                acc[i][j][r] = 0.0f;

    for (int st = 0; st < nstages; ++st) {
        const int p = st % NBUF;
#if SNN_TDM
        if (loader && st + 2 < nstages) stage_load(st + 2, (st + 2) % NBUF);
#else
        if (st + 1 < nstages) ldg((st + 1) * KSTEP);
#endif
        const __bf16* As = smem + p * SLAB;
        const __bf16* Ws = As + 128 * KSTEP;
#pragma unroll
        for (int ks = 0; ks < 2; ++ks) {
            v16bf af[4], bfr[2];
            const int ka = ks * 32 + hl * 8;   // A 16-bit 16x32 layout: K 0..7 / 16..23 per half
            const int kb = ks * 32 + hl * 8;   // B: same K order as A: 0..7 / 16..23 per half
#pragma unroll
            for (int j = 0; j < 2; ++j) {
                const int col = wn * 32 + j * 16 + l16;
                bfr[j] = frag16(Ws + col * KSTEP + kb, Ws + col * KSTEP + kb + 16);
            }
#pragma unroll
            for (int i = 0; i < 4; ++i) {
                const int row = wm * 64 + i * 16 + l16;
                af[i] = frag16(As + row * KSTEP + ka, As + row * KSTEP + ka + 16);
            }
#pragma unroll
            for (int i = 0; i < 4; ++i)
#pragma unroll
                for (int j = 0; j < 2; ++j)
                    acc[i][j] = __builtin_amdgcn_wmma_f32_16x16x32_bf16(
                        false, af[i], false, bfr[j], (short)0, acc[i][j], false, false);
        }
#if SNN_TDM
        // Ensure stage st+1 landed; leave stage st+2 (newest 2 tensor ops) in flight.
        if (st + 2 < nstages)      __builtin_amdgcn_s_wait_tensorcnt(2);
        else if (st + 1 < nstages) __builtin_amdgcn_s_wait_tensorcnt(0);
#else
        if (st + 1 < nstages) sts((st + 1) % NBUF);
#endif
        __syncthreads();
    }

    // ---- fused LIF epilogue (C/D layout: VGPR r -> row hl*8+r, col = l16) ----
#pragma unroll
    for (int i = 0; i < 4; ++i) {
#pragma unroll
        for (int j = 0; j < 2; ++j) {
            const int gn = bn0 + wn * 32 + j * 16 + l16;
            const float bv = bias[gn];
#pragma unroll
            for (int r = 0; r < 8; ++r) {
                const int gm = bm0 + wm * 64 + i * 16 + hl * 8 + r;
                const size_t idx = (size_t)gm * (size_t)N + gn;
                const float cur = acc[i][j][r] + bv;
                const float m = SNN_BETA * memv[idx] + cur - (float)spk[idx] * SNN_THRESH;
                memv[idx] = m;
                const float sp = (m - SNN_THRESH) > 0.0f ? 1.0f : 0.0f;
                spk[idx] = (__bf16)sp;
                if constexpr (ACCUM_OUT) out[idx] += sp;
            }
        }
    }
}

__global__ __launch_bounds__(256) void cvt_f32_bf16(
    const float* __restrict__ in, __bf16* __restrict__ out, size_t n8)
{
    size_t i = (size_t)blockIdx.x * blockDim.x + threadIdx.x;
    const size_t stride = (size_t)gridDim.x * blockDim.x;
    for (; i < n8; i += stride) {
        const float4* p = (const float4*)(in + i * 8);
        const float4 a = p[0], b = p[1];
        v8bf o;
        o[0] = (__bf16)a.x; o[1] = (__bf16)a.y; o[2] = (__bf16)a.z; o[3] = (__bf16)a.w;
        o[4] = (__bf16)b.x; o[5] = (__bf16)b.y; o[6] = (__bf16)b.z; o[7] = (__bf16)b.w;
        *(v8bf*)(out + i * 8) = o;
    }
}

extern "C" void kernel_launch(void* const* d_in, const int* in_sizes, int n_in,
                              void* d_out, int out_size, void* d_ws, size_t ws_size,
                              hipStream_t stream) {
    (void)in_sizes; (void)n_in; (void)ws_size;
    const float* x  = (const float*)d_in[0];
    const float* W1 = (const float*)d_in[1];
    const float* b1 = (const float*)d_in[2];
    const float* W2 = (const float*)d_in[3];
    const float* b2 = (const float*)d_in[4];
    const float* W3 = (const float*)d_in[5];
    const float* b3 = (const float*)d_in[6];
    float* out = (float*)d_out;

    const int B = 4096, T = 5, Din = 1024, Dh = 4096, Dout = 1024;

    char* ws = (char*)d_ws;
    size_t off = 0;
    auto wsalloc = [&](size_t bytes) -> void* {
        void* p = ws + off;
        off = (off + bytes + 255) & ~(size_t)255;
        return p;
    };
    __bf16* xb  = (__bf16*)wsalloc((size_t)B * T * Din * 2);
    __bf16* w1b = (__bf16*)wsalloc((size_t)Dh * Din * 2);
    __bf16* w2b = (__bf16*)wsalloc((size_t)Dh * Dh * 2);
    __bf16* w3b = (__bf16*)wsalloc((size_t)Dout * Dh * 2);
    __bf16* s1  = (__bf16*)wsalloc((size_t)B * Dh * 2);
    __bf16* s2  = (__bf16*)wsalloc((size_t)B * Dh * 2);
    __bf16* s3  = (__bf16*)wsalloc((size_t)B * Dout * 2);
    float*  m1  = (float*)wsalloc((size_t)B * Dh * 4);
    float*  m2  = (float*)wsalloc((size_t)B * Dh * 4);
    float*  m3  = (float*)wsalloc((size_t)B * Dout * 4);

    // f32 -> bf16 operand conversion (deterministic, part of every call)
    cvt_f32_bf16<<<2048, 256, 0, stream>>>(x,  xb,  (size_t)B * T * Din / 8);
    cvt_f32_bf16<<<1024, 256, 0, stream>>>(W1, w1b, (size_t)Dh * Din / 8);
    cvt_f32_bf16<<<2048, 256, 0, stream>>>(W2, w2b, (size_t)Dh * Dh  / 8);
    cvt_f32_bf16<<<1024, 256, 0, stream>>>(W3, w3b, (size_t)Dout * Dh / 8);

    // zero state + output (bf16 0x0000 == +0.0)
    hipMemsetAsync(s1, 0, (size_t)B * Dh * 2, stream);
    hipMemsetAsync(s2, 0, (size_t)B * Dh * 2, stream);
    hipMemsetAsync(s3, 0, (size_t)B * Dout * 2, stream);
    hipMemsetAsync(m1, 0, (size_t)B * Dh * 4, stream);
    hipMemsetAsync(m2, 0, (size_t)B * Dh * 4, stream);
    hipMemsetAsync(m3, 0, (size_t)B * Dout * 4, stream);
    hipMemsetAsync(out, 0, (size_t)out_size * 4, stream);

    const dim3 blk(256);
    const dim3 g12(Dh / 128,  B / 128);   // 32 x 32 blocks
    const dim3 g3 (Dout / 128, B / 128);  //  8 x 32 blocks

    for (int t = 0; t < T; ++t) {
        gemm_lif_kernel<false><<<g12, blk, 0, stream>>>(
            xb + (size_t)t * Din, T * Din, w1b, b1, m1, s1, nullptr, Din, Dh);
        gemm_lif_kernel<false><<<g12, blk, 0, stream>>>(
            s1, Dh, w2b, b2, m2, s2, nullptr, Dh, Dh);
        gemm_lif_kernel<true><<<g3, blk, 0, stream>>>(
            s2, Dh, w3b, b3, m3, s3, out, Dh, Dout);
    }
}
